// MoeFcTokensRl_86449101734486
// MI455X (gfx1250) — compile-verified, matrix operand order checked by reading, not run on hardware
//
#include <hip/hip_runtime.h>
#include <hip/hip_bf16.h>

// Problem constants (match reference)
#define BB 512
#define TT 2048
#define DD 32
#define EE 64
#define KK 8
#define OO 1024
#define FF 256  // KK*DD

#define LDS_PITCH 264   // bf16 elements per LDS row (256 + 8 pad -> bank stride 4)

typedef __bf16 bf16_t;
typedef __attribute__((ext_vector_type(16))) __bf16 v16bf;
typedef __attribute__((ext_vector_type(4)))  __bf16 v4bf;
typedef __attribute__((ext_vector_type(8)))  float  v8f;

#define NEG_INF (-3.0e38f)

// ---------------------------------------------------------------------------
// CDNA5 async global->LDS copy (ASYNCcnt path). 16 bytes per active lane.
// vdst = LDS byte offset (low 32 bits of generic pointer), vaddr = 64-bit GVA.
// ---------------------------------------------------------------------------
__device__ __forceinline__ void async_copy_b128(void* lds_ptr, const void* gptr) {
    unsigned lds_off = (unsigned)(unsigned long long)lds_ptr;
    asm volatile("global_load_async_to_lds_b128 %0, %1, off"
                 :: "v"(lds_off), "v"(gptr)
                 : "memory");
}

__device__ __forceinline__ void wait_async_zero() {
    asm volatile("s_wait_asynccnt 0x0" ::: "memory");
}

// ---------------------------------------------------------------------------
// Kernel A: expert_w fp32 -> bf16 (packed)
// ---------------------------------------------------------------------------
__global__ __launch_bounds__(256)
void cvt_w_kernel(const float* __restrict__ w, bf16_t* __restrict__ o, int n4) {
    int i = blockIdx.x * 256 + threadIdx.x;
    if (i >= n4) return;
    float4 f = ((const float4*)w)[i];
    v4bf r = { (bf16_t)f.x, (bf16_t)f.y, (bf16_t)f.z, (bf16_t)f.w };
    ((v4bf*)o)[i] = r;
}

// ---------------------------------------------------------------------------
// Kernel B: gating GEMV + log_softmax over tokens + Gumbel top-8 + gather
// One wave32 per (b,e) instance; 8 waves / block; 64KB static LDS.
// ---------------------------------------------------------------------------
__global__ __launch_bounds__(256)
void moe_gate_kernel(const float* __restrict__ x,
                     const float* __restrict__ gate_w,
                     const float* __restrict__ gate_b,
                     const float* __restrict__ gumbel,
                     bf16_t* __restrict__ Ab,
                     int* __restrict__ idx_out) {
    __shared__ float s_sc[8 * TT];   // per-wave 2048-score slice

    const int lane = threadIdx.x & 31;
    const int wid  = threadIdx.x >> 5;
    const int inst = blockIdx.x * 8 + wid;   // (b,e) instance
    const int b = inst / EE;
    const int e = inst % EE;
    float* sc = s_sc + wid * TT;

    // preload gate weights for expert e (32 floats)
    float4 g[8];
    const float4* gw4 = (const float4*)(gate_w + e * DD);
#pragma unroll
    for (int c = 0; c < 8; ++c) g[c] = gw4[c];
    const float bias_e = gate_b[e];

    const float* xb = x + (size_t)b * TT * DD;

    // pass 1: logits -> LDS, track local max
    float lmax = NEG_INF;
    for (int j = 0; j < 64; ++j) {
        int t = j * 32 + lane;
        const float4* xv = (const float4*)(xb + (size_t)t * DD);
        float acc = bias_e;
#pragma unroll
        for (int c = 0; c < 8; ++c) {
            float4 v = xv[c];
            acc += v.x * g[c].x + v.y * g[c].y + v.z * g[c].z + v.w * g[c].w;
        }
        sc[t] = acc;
        lmax = fmaxf(lmax, acc);
    }
#pragma unroll
    for (int off = 16; off; off >>= 1)
        lmax = fmaxf(lmax, __shfl_xor(lmax, off, 32));

    // pass 2: logsumexp over tokens
    float s = 0.0f;
    for (int j = 0; j < 64; ++j) {
        int t = j * 32 + lane;
        s += __expf(sc[t] - lmax);
    }
#pragma unroll
    for (int off = 16; off; off >>= 1)
        s += __shfl_xor(s, off, 32);
    const float lse = lmax + __logf(s);

    // pass 3: score = logp + gumbel
    const float* gmb = gumbel + (size_t)inst * TT;
    for (int j = 0; j < 64; ++j) {
        int t = j * 32 + lane;
        sc[t] = (sc[t] - lse) + gmb[t];
    }
    __syncthreads();

    // iterative top-8 (descending, low-index tie-break like jax top_k)
    int topt[KK];
#pragma unroll
    for (int k = 0; k < KK; ++k) {
        float bv = NEG_INF;
        int   bt = TT;
        for (int j = 0; j < 64; ++j) {
            int t = j * 32 + lane;
            float vv = sc[t];
            if (vv > bv || (vv == bv && t < bt)) { bv = vv; bt = t; }
        }
#pragma unroll
        for (int off = 16; off; off >>= 1) {
            float ov = __shfl_xor(bv, off, 32);
            int   ot = __shfl_xor(bt, off, 32);
            if (ov > bv || (ov == bv && ot < bt)) { bv = ov; bt = ot; }
        }
        topt[k] = bt;                         // uniform across wave
        if (lane == (bt & 31)) sc[bt] = NEG_INF;
        __syncthreads();
    }

    if (lane == 0) {
#pragma unroll
        for (int k = 0; k < KK; ++k) idx_out[inst * KK + k] = topt[k];
    }

    // gather 8 tokens x 32 feats, convert to bf16 : Ab[inst][k*32 + d]
    bf16_t* dst = Ab + (size_t)inst * FF;
#pragma unroll
    for (int k = 0; k < KK; ++k) {
        float xv = xb[(size_t)topt[k] * DD + lane];
        dst[k * 32 + lane] = (bf16_t)xv;
    }
}

// ---------------------------------------------------------------------------
// Kernel C: grouped GEMM via v_wmma_f32_16x16x32_bf16, operands staged in LDS
// with global_load_async_to_lds_b128 (ASYNCcnt).
// out[b,e,o] = sum_f Ab[b,e,f] * Wb[e,o,f] + bias[e,o]
// Block: 256 thr (8 waves). Block tile: M=64 (batch), N=128 (OUT), one expert.
// Dynamic LDS: A tile 64x264 bf16 + W tile 128x264 bf16 (~99 KB of 320 KB WGP).
// Wave: m_sub = wid&3 (16 rows), n range = (wid>>2)*64 .. +63 (4 tiles).
// ---------------------------------------------------------------------------
__device__ __forceinline__ v16bf lds_frag16(const bf16_t* base, int half) {
    // ISA 16-bit A/B layout: half 0 holds K 0..7 then 16..23; half 1 holds
    // K 8..15 then 24..31 -> two contiguous 16B chunks per lane.
    union { uint4 u[2]; v16bf v; } f;
    f.u[0] = *(const uint4*)(base + half * 8);
    f.u[1] = *(const uint4*)(base + 16 + half * 8);
    return f.v;
}

__global__ __launch_bounds__(256)
void moe_gemm_kernel(const bf16_t* __restrict__ A,
                     const bf16_t* __restrict__ W,
                     const float* __restrict__ bias,
                     float* __restrict__ out) {
    extern __shared__ bf16_t smem[];
    bf16_t* sA = smem;                    // [64][LDS_PITCH]
    bf16_t* sW = smem + 64 * LDS_PITCH;   // [128][LDS_PITCH]

    const int tid  = threadIdx.x;
    const int lane = tid & 31;
    const int wid  = tid >> 5;
    const int half = lane >> 4;
    const int nl   = lane & 15;
    const int e    = blockIdx.z;
    const int m_blk = blockIdx.y * 64;
    const int n_blk = blockIdx.x * 128;

    // ---- async stage A tile: 64 rows x 32 chunks of 16B -------------------
#pragma unroll
    for (int it = 0; it < 8; ++it) {
        int c = tid + it * 256;           // 0 .. 2047
        int row = c >> 5, col = c & 31;
        const bf16_t* g = A + ((size_t)(m_blk + row) * EE + e) * FF + col * 8;
        async_copy_b128(sA + row * LDS_PITCH + col * 8, g);
    }
    // ---- async stage W tile: 128 rows x 32 chunks of 16B ------------------
#pragma unroll
    for (int it = 0; it < 16; ++it) {
        int c = tid + it * 256;           // 0 .. 4095
        int row = c >> 5, col = c & 31;
        const bf16_t* g = W + ((size_t)e * OO + n_blk + row) * FF + col * 8;
        async_copy_b128(sW + row * LDS_PITCH + col * 8, g);
    }
    wait_async_zero();
    __syncthreads();

    // ---- WMMA compute from LDS -------------------------------------------
    const bf16_t* a_row = sA + ((wid & 3) * 16 + nl) * LDS_PITCH;
    const bf16_t* w_base = sW + ((wid >> 2) * 64 + nl) * LDS_PITCH;

    v8f acc[4] = {};
#pragma unroll
    for (int ks = 0; ks < 8; ++ks) {
        const int kb = ks * 32;
        v16bf af = lds_frag16(a_row + kb, half);
#pragma unroll
        for (int j = 0; j < 4; ++j) {
            v16bf bfrag = lds_frag16(w_base + j * 16 * LDS_PITCH + kb, half);
            acc[j] = __builtin_amdgcn_wmma_f32_16x16x32_bf16(
                false, af, false, bfrag, (short)0, acc[j], false, false);
        }
    }

    // ---- epilogue: bias + store (out is [B,E,OUT] fp32) -------------------
    const int m_base = m_blk + (wid & 3) * 16;
    const int n_base = n_blk + (wid >> 2) * 64;
#pragma unroll
    for (int j = 0; j < 4; ++j) {
        const int o  = n_base + j * 16 + nl;
        const float bv = bias[e * OO + o];
#pragma unroll
        for (int v = 0; v < 8; ++v) {
            const int m = m_base + v + 8 * half;  // C/D layout: M = v + 8*half
            out[((size_t)m * EE + e) * OO + o] = acc[j][v] + bv;
        }
    }
}

// ---------------------------------------------------------------------------
// Launch
// ---------------------------------------------------------------------------
extern "C" void kernel_launch(void* const* d_in, const int* in_sizes, int n_in,
                              void* d_out, int out_size, void* d_ws, size_t ws_size,
                              hipStream_t stream) {
    (void)in_sizes; (void)n_in; (void)out_size; (void)ws_size;

    const float* x        = (const float*)d_in[0];
    const float* gate_w   = (const float*)d_in[1];
    const float* gate_b   = (const float*)d_in[2];
    const float* expert_w = (const float*)d_in[3];
    const float* expert_b = (const float*)d_in[4];
    const float* gumbel   = (const float*)d_in[5];
    float* out = (float*)d_out;

    // workspace layout: [W bf16 | A bf16 | idx i32]
    const size_t w_bytes = (size_t)EE * OO * FF * sizeof(bf16_t);  // 33.5 MB
    const size_t a_bytes = (size_t)BB * EE * FF * sizeof(bf16_t);  // 16.8 MB
    bf16_t* Wb  = (bf16_t*)d_ws;
    bf16_t* Ab  = (bf16_t*)((char*)d_ws + w_bytes);
    int*    idx = (int*)((char*)d_ws + w_bytes + a_bytes);

    const int n4 = EE * OO * FF / 4;
    cvt_w_kernel<<<(n4 + 255) / 256, 256, 0, stream>>>(expert_w, Wb, n4);

    moe_gate_kernel<<<BB * EE / 8, 256, 0, stream>>>(
        x, gate_w, gate_b, gumbel, Ab, idx);

    const size_t smem_bytes = (size_t)(64 + 128) * LDS_PITCH * sizeof(bf16_t);
    moe_gemm_kernel<<<dim3(OO / 128, BB / 64, EE), 256, smem_bytes, stream>>>(
        Ab, Wb, expert_b, out);
}
